// GATmannEncoder_52475910422750
// MI455X (gfx1250) — matrix operand order repaired and checked by reading, not compiled, not run on hardware
//
#include <hip/hip_runtime.h>
#include <hip/hip_bf16.h>

typedef __attribute__((ext_vector_type(16))) _Float16 v16h;
typedef __attribute__((ext_vector_type(8)))  _Float16 v8h;
typedef __attribute__((ext_vector_type(8)))  float    v8f;

#define F_NODE 79
#define F_EDGE 10
#define NEG_ATT 0.2f
#define NEG_ACT 0.01f

// -------- float atomic max via int trick (init with -inf) --------
__device__ __forceinline__ void atomicMaxF(float* addr, float v) {
  if (v >= 0.0f) atomicMax((int*)addr, __float_as_int(v));
  else           atomicMin((unsigned int*)addr, __float_as_uint(v));
}

__global__ void fill_kernel(float* __restrict__ p, float v, int n) {
  int i = blockIdx.x * blockDim.x + threadIdx.x;
  if (i < n) p[i] = v;
}

// Ah[Mp, KP] (f16) = leaky(A[M, K], slope), zero-padded in K and rows
__global__ void prep_a_kernel(const float* __restrict__ A, _Float16* __restrict__ Ah,
                              int M, int K, int KP, int Mp, float slope)
{
  int i = blockIdx.x * blockDim.x + threadIdx.x;
  if (i >= Mp * KP) return;
  int row = i / KP, k = i % KP;
  float v = (row < M && k < K) ? A[row * K + k] : 0.0f;
  v = (v >= 0.0f) ? v : v * slope;
  Ah[i] = (_Float16)v;
}

// Wt[2*HC, KP] (f16, col-major weights): row c<HC -> Wl[:,c]; row c>=HC -> Wr[:,c-HC]
__global__ void prep_w_kernel(const float* __restrict__ Wl, const float* __restrict__ Wr,
                              _Float16* __restrict__ Wt, int K, int KP, int HC)
{
  int i = blockIdx.x * blockDim.x + threadIdx.x;
  if (i >= 2 * HC * KP) return;
  int col = i / KP, k = i % KP;
  float v = 0.0f;
  if (k < K) v = (col < HC) ? Wl[k * HC + col] : Wr[k * HC + (col - HC)];
  Wt[i] = (_Float16)v;
}

// C[Mp, NC2] = Ah[Mp, KP] @ Wt^T + packed bias. One wave per 16x16 tile; fully
// unrolled v_wmma_f32_16x16x32_f16 chain with b128 operand loads and
// guard-free constant-offset stores (C is row-padded to Mp).
template <int KP, int NC2>
__global__ void gemm_wmma_kernel(const _Float16* __restrict__ Ah,
                                 const _Float16* __restrict__ Wt,
                                 const float* __restrict__ bl,
                                 const float* __restrict__ br,
                                 float* __restrict__ C, int Mp)
{
  constexpr int HC = NC2 / 2;
  int gtid    = blockIdx.x * blockDim.x + threadIdx.x;
  int wave    = gtid >> 5;                 // uniform across the wave
  int lane    = threadIdx.x & 31;
  int tiles_m = Mp >> 4;
  int tiles_n = NC2 >> 4;
  if (wave >= tiles_m * tiles_n) return;   // wave-uniform exit
  int tm = wave % tiles_m;
  int tn = wave / tiles_m;
  int m0 = tm << 4, n0 = tn << 4;
  int r  = lane & 15;
  int hi = lane >> 4;
  const _Float16* arow = Ah + (m0 + r) * KP;   // this lane's A row
  const _Float16* bcol = Wt + (n0 + r) * KP;   // this lane's W column

  v8f acc = {};
#pragma unroll
  for (int kb = 0; kb < KP; kb += 32) {
    // A fragment: j<8 -> K=kb+8*hi+j ; j>=8 -> K=kb+16+8*hi+(j-8)
    v8h a0 = *(const v8h*)(arow + kb + 8 * hi);
    v8h a1 = *(const v8h*)(arow + kb + 16 + 8 * hi);
    // B fragment: j<8 -> K=kb+8*hi+j ; j>=8 -> K=kb+16+8*hi+(j-8)
    v8h b0 = *(const v8h*)(bcol + kb + 8 * hi);
    v8h b1 = *(const v8h*)(bcol + kb + 16 + 8 * hi);
    v16h a = __builtin_shufflevector(a0, a1, 0,1,2,3,4,5,6,7,8,9,10,11,12,13,14,15);
    v16h b = __builtin_shufflevector(b0, b1, 0,1,2,3,4,5,6,7,8,9,10,11,12,13,14,15);
    acc = __builtin_amdgcn_wmma_f32_16x16x32_f16(false, a, false, b,
                                                 (short)0, acc, false, false);
  }
  int col = n0 + r;
  float bc = (col < HC) ? bl[col] : br[col - HC];
  float* cp = C + (m0 + 8 * hi) * NC2 + col;   // rows m0+8*hi .. +7
#pragma unroll
  for (int i = 0; i < 8; ++i) cp[i * NC2] = acc[i] + bc;
}

// Per (edge, head): logit = sum_c leaky(xl[s]+xr[d]+ef, 0.2) . att ; segment-max into mx
// xlr layout: [Mp, 2*HC]; xl at offset 0, xr at offset HC.
__global__ void edge_logits_kernel(const int* __restrict__ src, const int* __restrict__ dst,
                                   const float* __restrict__ edge_attr,
                                   const float* __restrict__ We, const float* __restrict__ att,
                                   const float* __restrict__ xlr,
                                   float* __restrict__ logits, float* __restrict__ mx,
                                   int E_, int H, int Cc, int HC)
{
  int idx = blockIdx.x * blockDim.x + threadIdx.x;
  if (idx >= E_ * H) return;
  int e = idx / H, h = idx % H;
  int s = src[e], d = dst[e];
  int S = 2 * HC;
  float ea[F_EDGE];
#pragma unroll
  for (int f = 0; f < F_EDGE; ++f) ea[f] = edge_attr[e * F_EDGE + f];
  const float* xls  = xlr + (size_t)s * S + h * Cc;
  const float* xrd  = xlr + (size_t)d * S + HC + h * Cc;
  const float* Weh  = We + h * Cc;
  const float* atth = att + h * Cc;
  float logit = 0.0f;
  for (int c = 0; c < Cc; ++c) {
    float ef = 0.0f;
#pragma unroll
    for (int f = 0; f < F_EDGE; ++f) ef += ea[f] * Weh[f * HC + c];
    float m = xls[c] + xrd[c] + ef;
    m = (m >= 0.0f) ? m : m * NEG_ATT;
    logit += m * atth[c];
  }
  logits[idx] = logit;
  atomicMaxF(mx + d * H + h, logit);
}

// Per (edge, head): ex = exp(logit - mx[dst]); segment-sum into denom; overwrite logits with ex
__global__ void edge_exp_kernel(const int* __restrict__ dst, float* __restrict__ logits,
                                const float* __restrict__ mx, float* __restrict__ denom,
                                int E_, int H)
{
  int idx = blockIdx.x * blockDim.x + threadIdx.x;
  if (idx >= E_ * H) return;
  int e = idx / H, h = idx % H;
  int d = dst[e];
  float ex = __expf(logits[idx] - mx[d * H + h]);
  logits[idx] = ex;
  atomicAdd(denom + d * H + h, ex);
}

// Per (edge, head): accum[dst, h, :] += xl[src, h, :] * (ex / denom[dst, h])
__global__ void edge_scatter_kernel(const int* __restrict__ src, const int* __restrict__ dst,
                                    const float* __restrict__ ex, const float* __restrict__ denom,
                                    const float* __restrict__ xlr, float* __restrict__ accum,
                                    int E_, int H, int Cc, int HC)
{
  int idx = blockIdx.x * blockDim.x + threadIdx.x;
  if (idx >= E_ * H) return;
  int e = idx / H, h = idx % H;
  int s = src[e], d = dst[e];
  float alpha = ex[idx] / fmaxf(denom[d * H + h], 1e-16f);
  const float* xls = xlr + (size_t)s * 2 * HC + h * Cc;
  float* acc = accum + (size_t)d * HC + h * Cc;
  for (int c = 0; c < Cc; ++c) atomicAdd(acc + c, xls[c] * alpha);
}

__global__ void finalize_concat_kernel(const float* __restrict__ accum, const float* __restrict__ bias,
                                       float* __restrict__ out, int n_, int w)
{
  int i = blockIdx.x * blockDim.x + threadIdx.x;
  if (i >= n_ * w) return;
  out[i] = accum[i] + bias[i % w];
}

__global__ void finalize_mean_kernel(const float* __restrict__ accum, const float* __restrict__ bias,
                                     float* __restrict__ out, int n_, int Cc)
{
  int i = blockIdx.x * blockDim.x + threadIdx.x;
  if (i >= n_ * Cc) return;
  int n = i / Cc, c = i % Cc;
  out[i] = 0.5f * (accum[n * 2 * Cc + c] + accum[n * 2 * Cc + Cc + c]) + bias[c];
}

static inline int cdiv(int a, int b) { return (a + b - 1) / b; }

extern "C" void kernel_launch(void* const* d_in, const int* in_sizes, int n_in,
                              void* d_out, int out_size, void* d_ws, size_t ws_size,
                              hipStream_t stream)
{
  const float* x         = (const float*)d_in[0];
  const int*   edge_idx  = (const int*)  d_in[1];
  const float* edge_attr = (const float*)d_in[2];
  const float* Wl0  = (const float*)d_in[3];
  const float* bl0  = (const float*)d_in[4];
  const float* Wr0  = (const float*)d_in[5];
  const float* br0  = (const float*)d_in[6];
  const float* We0  = (const float*)d_in[7];
  const float* att0 = (const float*)d_in[8];
  const float* bias0= (const float*)d_in[9];
  const float* Wl1  = (const float*)d_in[10];
  const float* bl1  = (const float*)d_in[11];
  const float* Wr1  = (const float*)d_in[12];
  const float* br1  = (const float*)d_in[13];
  const float* We1  = (const float*)d_in[14];
  const float* att1 = (const float*)d_in[15];
  const float* bias1= (const float*)d_in[16];

  const int N  = in_sizes[0] / F_NODE;
  const int E  = in_sizes[1] / 2;
  const int Mp = cdiv(N, 16) * 16;
  const int* src = edge_idx;
  const int* dst = edge_idx + E;
  float* out = (float*)d_out;

  // workspace layout (16B-aligned sections)
  float* ws     = (float*)d_ws;
  float* xlr    = ws;                            // Mp * 256 (max 2*HC), row-padded
  float* accum  = xlr + (size_t)Mp * 256;        // N * 128
  float* h      = accum + (size_t)N * 128;       // N * 64
  float* logits = h + (size_t)N * 64;            // E * 2
  float* mx     = logits + (size_t)E * 2;        // N * 2
  float* denom  = mx + (size_t)N * 2;            // N * 2
  _Float16* Ah  = (_Float16*)(denom + (size_t)N * 2);  // Mp * 96 halves (max)
  _Float16* Wt  = Ah + (size_t)Mp * 96;                // 256 * 96 halves

  const int TB = 256;

  for (int layer = 0; layer < 3; ++layer) {
    const bool first = (layer == 0);
    const int  li    = first ? 0 : (layer - 1);
    const int  K     = first ? F_NODE : 64;
    const int  KP    = first ? 96 : 64;          // K padded to multiple of 32
    const int  HC    = first ? 64 : 128;         // per-matrix output width (H*C)
    const int  Cc    = first ? 32 : 64;
    const int  H     = 2;
    const float slope = first ? 1.0f : NEG_ACT;
    const float* Ain  = first ? x : h;
    const float* Wl = first ? Wl0 : (Wl1 + li * 64 * 128);
    const float* bl = first ? bl0 : (bl1 + li * 128);
    const float* Wr = first ? Wr0 : (Wr1 + li * 64 * 128);
    const float* br = first ? br0 : (br1 + li * 128);
    const float* We = first ? We0 : (We1 + li * F_EDGE * 128);
    const float* at = first ? att0 : (att1 + li * 2 * 64);
    const float* bs = first ? bias0 : (bias1 + li * 64);

    // operand prep: f16 conversion + padding + weight transpose/packing
    prep_a_kernel<<<cdiv(Mp * KP, TB), TB, 0, stream>>>(Ain, Ah, N, K, KP, Mp, slope);
    prep_w_kernel<<<cdiv(2 * HC * KP, TB), TB, 0, stream>>>(Wl, Wr, Wt, K, KP, HC);

    // init segment buffers
    fill_kernel<<<cdiv(N * H, TB), TB, 0, stream>>>(mx, -INFINITY, N * H);
    fill_kernel<<<cdiv(N * H, TB), TB, 0, stream>>>(denom, 0.0f, N * H);
    fill_kernel<<<cdiv(N * HC, TB), TB, 0, stream>>>(accum, 0.0f, N * HC);

    // fused node transform: xlr = leaky(A) @ [Wl | Wr] + [bl | br]  (WMMA)
    int tiles   = (Mp / 16) * (2 * HC / 16);
    int gblocks = cdiv(tiles * 32, 128);
    if (first) gemm_wmma_kernel<96, 128><<<gblocks, 128, 0, stream>>>(Ah, Wt, bl, br, xlr, Mp);
    else       gemm_wmma_kernel<64, 256><<<gblocks, 128, 0, stream>>>(Ah, Wt, bl, br, xlr, Mp);

    // edge attention pipeline
    int eh = E * H;
    edge_logits_kernel<<<cdiv(eh, TB), TB, 0, stream>>>(src, dst, edge_attr, We, at,
                                                        xlr, logits, mx, E, H, Cc, HC);
    edge_exp_kernel<<<cdiv(eh, TB), TB, 0, stream>>>(dst, logits, mx, denom, E, H);
    edge_scatter_kernel<<<cdiv(eh, TB), TB, 0, stream>>>(src, dst, logits, denom, xlr,
                                                         accum, E, H, Cc, HC);

    // finalize
    if (first) {
      finalize_concat_kernel<<<cdiv(N * 64, TB), TB, 0, stream>>>(accum, bs, h, N, 64);
    } else {
      float* dest = (layer == 1) ? h : out;
      finalize_mean_kernel<<<cdiv(N * 64, TB), TB, 0, stream>>>(accum, bs, dest, N, 64);
    }
  }
}
